// MultiHeadAttention_6536940224773
// MI455X (gfx1250) — compile-verified, matrix operand order checked by reading, not run on hardware
//
#include <hip/hip_runtime.h>

// ---------------------------------------------------------------------------
// MI455X (gfx1250) wave32 WMMA implementation of causal multi-head attention.
//   B=2, S=2048, D_IN=1024, D_OUT=1024, H=16, HD=64
// Pipeline:
//   1) qkv_gemm : x(f32) @ [Wq|Wk|Wv](f32) -> Q,K (f16 [b,h,s,hd]), Vt (f16 [b,h,hd,s])
//   2) attn     : flash-style causal attention per 16-query wave tile -> ctx f16
//   3) out_gemm : ctx(f16) @ Wo(f32->f16) + bo -> out f32
// All matrix math uses v_wmma_f32_16x16x32_f16.
// ---------------------------------------------------------------------------

typedef __attribute__((ext_vector_type(16))) _Float16 v16h;
typedef __attribute__((ext_vector_type(8)))  _Float16 v8h;
typedef __attribute__((ext_vector_type(4)))  _Float16 v4h;
typedef __attribute__((ext_vector_type(8)))  float    v8f;

#define BDIM_B   2
#define SEQ      2048
#define DIN      1024
#define DOUT     1024
#define NHEAD    16
#define HDIM     64
#define QTILES   (SEQ / 16)        // 128 query tiles per (b,h)

// 16-bit A/B fragment loader (CDNA5 ISA 7.12.2 layout):
// lane holds row/col (lane&15); K elements come as two contiguous 8-halfword
// chunks at K = (lane>>4)*8 and K = 16 + (lane>>4)*8 within the 32-wide K tile.
__device__ __forceinline__ v16h frag_ld(const _Float16* rowBase, int halfSel) {
  const v8h lo = *(const v8h*)(rowBase + halfSel * 8);
  const v8h hi = *(const v8h*)(rowBase + 16 + halfSel * 8);
  return __builtin_shufflevector(lo, hi, 0, 1, 2, 3, 4, 5, 6, 7,
                                 8, 9, 10, 11, 12, 13, 14, 15);
}

__device__ __forceinline__ v8f wmma16(v16h a, v16h b, v8f c) {
  // (neg_a, A, neg_b, B, c_mod, C, reuse_a, reuse_b)
  return __builtin_amdgcn_wmma_f32_16x16x32_f16(false, a, false, b,
                                                (short)0, c, false, false);
}

// ---------------------------------------------------------------------------
// Kernel 1: fused QKV projection.  Grid (M/128, 3072/64), 256 threads (8 waves)
// ---------------------------------------------------------------------------
#define LDSK 40   // padded f16 stride for 32-wide K tiles (80B = 5*16B aligned)

__global__ __launch_bounds__(256) void qkv_gemm(
    const float* __restrict__ x, const float* __restrict__ Wq,
    const float* __restrict__ Wk, const float* __restrict__ Wv,
    _Float16* __restrict__ Qb, _Float16* __restrict__ Kb,
    _Float16* __restrict__ Vtb) {
  __shared__ _Float16 As[128 * LDSK];
  __shared__ _Float16 Bs[64 * LDSK];

  const int tid  = threadIdx.x;
  const int lane = tid & 31;
  const int w    = tid >> 5;
  const int lh   = lane & 15;
  const int hs   = lane >> 4;
  const int wm   = (w & 3) * 32;
  const int wn   = (w >> 2) * 32;
  const int m0   = blockIdx.x * 128;
  const int n0   = blockIdx.y * 64;          // 0..3071
  const int mat  = n0 >> 10;                 // 0:Q 1:K 2:V
  const int nc0  = n0 & 1023;
  const float* __restrict__ W = (mat == 0) ? Wq : ((mat == 1) ? Wk : Wv);

  v8f c[2][2] = {};

  for (int k0 = 0; k0 < DIN; k0 += 32) {
    // ---- stage A (x) tile 128x32 fp32 -> f16 LDS ----
#pragma unroll
    for (int p = 0; p < 4; ++p) {
      const int r  = (tid >> 3) + p * 32;
      const int c4 = (tid & 7) * 4;
      const float4 v = *(const float4*)(x + (size_t)(m0 + r) * DIN + k0 + c4);
      v4h hv = {(_Float16)v.x, (_Float16)v.y, (_Float16)v.z, (_Float16)v.w};
      *(v4h*)(As + r * LDSK + c4) = hv;
    }
    // ---- stage B (W) tile 32x64 fp32 -> transposed f16 LDS Bt[n][k] ----
#pragma unroll
    for (int p = 0; p < 2; ++p) {
      const int kk = (tid >> 4) + p * 16;
      const int nn = (tid & 15) * 4;
      const float4 v = *(const float4*)(W + (size_t)(k0 + kk) * DOUT + nc0 + nn);
      Bs[(nn + 0) * LDSK + kk] = (_Float16)v.x;
      Bs[(nn + 1) * LDSK + kk] = (_Float16)v.y;
      Bs[(nn + 2) * LDSK + kk] = (_Float16)v.z;
      Bs[(nn + 3) * LDSK + kk] = (_Float16)v.w;
    }
    if (k0 + 32 < DIN) {  // next-tile prefetch -> global_prefetch_b8
      __builtin_prefetch(x + (size_t)(m0 + (tid >> 3)) * DIN + k0 + 32, 0, 0);
      __builtin_prefetch(W + (size_t)(k0 + 32 + (tid >> 4)) * DOUT + nc0, 0, 0);
    }
    __syncthreads();

    const v16h a0 = frag_ld(As + (wm + 0 + lh) * LDSK, hs);
    const v16h a1 = frag_ld(As + (wm + 16 + lh) * LDSK, hs);
    const v16h b0 = frag_ld(Bs + (wn + 0 + lh) * LDSK, hs);
    const v16h b1 = frag_ld(Bs + (wn + 16 + lh) * LDSK, hs);
    c[0][0] = wmma16(a0, b0, c[0][0]);
    c[0][1] = wmma16(a0, b1, c[0][1]);
    c[1][0] = wmma16(a1, b0, c[1][0]);
    c[1][1] = wmma16(a1, b1, c[1][1]);
    __syncthreads();
  }

  // ---- epilogue: scatter f16 into Q / K / Vt workspace layouts ----
#pragma unroll
  for (int i = 0; i < 2; ++i) {
#pragma unroll
    for (int j = 0; j < 2; ++j) {
#pragma unroll
      for (int r = 0; r < 8; ++r) {
        const int gi = m0 + wm + i * 16 + hs * 8 + r;   // row in [0, B*S)
        const int gj = n0 + wn + j * 16 + lh;           // col in [0, 3*DOUT)
        const int b  = gi >> 11;                        // /S
        const int s  = gi & (SEQ - 1);
        const int d  = gj & (DOUT - 1);
        const int h  = d >> 6;                          // /HD
        const int hd = d & (HDIM - 1);
        const _Float16 val = (_Float16)c[i][j][r];
        if (mat == 0) {
          Qb[((size_t)(b * NHEAD + h) * SEQ + s) * HDIM + hd] = val;
        } else if (mat == 1) {
          Kb[((size_t)(b * NHEAD + h) * SEQ + s) * HDIM + hd] = val;
        } else {
          Vtb[((size_t)(b * NHEAD + h) * HDIM + hd) * SEQ + s] = val;
        }
      }
    }
  }
}

// ---------------------------------------------------------------------------
// Kernel 2: flash-style causal attention. One wave per 16-query tile.
// Grid: B*H*QTILES / 8 waves = 512 blocks of 256 threads.
// ---------------------------------------------------------------------------
__global__ __launch_bounds__(256) void attn_kernel(
    const _Float16* __restrict__ Qb, const _Float16* __restrict__ Kb,
    const _Float16* __restrict__ Vtb, _Float16* __restrict__ Ctx) {
  __shared__ _Float16 P[8][16 * 32];   // per-wave f16 probability tile

  const int tid  = threadIdx.x;
  const int lane = tid & 31;
  const int w    = tid >> 5;
  const int lh   = lane & 15;
  const int hs   = lane >> 4;
  const int gt   = blockIdx.x * 8 + w;
  const int qt   = gt & (QTILES - 1);  // query tile within (b,h)
  const int bh   = gt >> 7;            // 0..31

  // Q fragments for this 16-row query tile (K-dim = hd, two 32-chunks)
  const _Float16* qrow = Qb + (size_t)(bh * SEQ + qt * 16 + lh) * HDIM;
  const v16h qa0 = frag_ld(qrow, hs);
  const v16h qa1 = frag_ld(qrow + 32, hs);

  float mrow[8], lrow[8];
  v8f acc[4] = {};
#pragma unroll
  for (int r = 0; r < 8; ++r) { mrow[r] = -1.0e30f; lrow[r] = 0.0f; }

  const int qmax = qt * 16 + 15;
  _Float16* Pw = P[w];

  for (int kb = 0; kb * 32 <= qmax; ++kb) {
    const int key0 = kb * 32;

    // ---- scores S = Q @ K^T for 16 q x 32 keys (K-dim hd=64) ----
    v8f s0 = {}, s1 = {};
    {
      const _Float16* k0row = Kb + (size_t)(bh * SEQ + key0 + lh) * HDIM;
      const _Float16* k1row = k0row + 16 * HDIM;
      const v16h b00 = frag_ld(k0row, hs);
      const v16h b01 = frag_ld(k0row + 32, hs);
      const v16h b10 = frag_ld(k1row, hs);
      const v16h b11 = frag_ld(k1row + 32, hs);
      s0 = wmma16(qa0, b00, s0);
      s0 = wmma16(qa1, b01, s0);
      s1 = wmma16(qa0, b10, s1);
      s1 = wmma16(qa1, b11, s1);
    }

    // ---- online softmax (row stats live in 16 lanes of each C-frag row) ----
#pragma unroll
    for (int r = 0; r < 8; ++r) {
      const int row = qt * 16 + hs * 8 + r;   // absolute query index
      const int c0  = key0 + lh;
      const int c1  = key0 + 16 + lh;
      float e0 = (c0 <= row) ? s0[r] * 0.125f : -1.0e30f;  // 1/sqrt(64)
      float e1 = (c1 <= row) ? s1[r] * 0.125f : -1.0e30f;
      float mx = fmaxf(e0, e1);
      mx = fmaxf(mx, __shfl_xor(mx, 1));
      mx = fmaxf(mx, __shfl_xor(mx, 2));
      mx = fmaxf(mx, __shfl_xor(mx, 4));
      mx = fmaxf(mx, __shfl_xor(mx, 8));
      const float mn = fmaxf(mrow[r], mx);
      const float p0 = __expf(e0 - mn);
      const float p1 = __expf(e1 - mn);
      float rs = p0 + p1;
      rs += __shfl_xor(rs, 1);
      rs += __shfl_xor(rs, 2);
      rs += __shfl_xor(rs, 4);
      rs += __shfl_xor(rs, 8);
      const float sc = __expf(mrow[r] - mn);
      lrow[r] = lrow[r] * sc + rs;
      mrow[r] = mn;
#pragma unroll
      for (int j = 0; j < 4; ++j) acc[j][r] *= sc;
      const int rl = hs * 8 + r;
      Pw[rl * 32 + lh]      = (_Float16)p0;
      Pw[rl * 32 + 16 + lh] = (_Float16)p1;
    }

    // ---- ctx += P @ V  (K-dim = 32 keys; Vt gives contiguous key chunks) ----
    const v16h pa = frag_ld(Pw + lh * 32, hs);
#pragma unroll
    for (int j = 0; j < 4; ++j) {
      const _Float16* vrow =
          Vtb + (size_t)(bh * HDIM + j * 16 + lh) * SEQ + key0;
      const v16h vb = frag_ld(vrow, hs);
      acc[j] = wmma16(pa, vb, acc[j]);
    }
  }

  // ---- epilogue: normalize and store ctx f16 as [b, s, h*64+hd] ----
  const int b = bh >> 4;
  const int h = bh & 15;
#pragma unroll
  for (int j = 0; j < 4; ++j) {
#pragma unroll
    for (int r = 0; r < 8; ++r) {
      const int rl = hs * 8 + r;
      const int s  = qt * 16 + rl;
      const float val = acc[j][r] / lrow[r];
      Ctx[(size_t)(b * SEQ + s) * DOUT + h * HDIM + j * 16 + lh] =
          (_Float16)val;
    }
  }
}

// ---------------------------------------------------------------------------
// Kernel 3: output projection  out = ctx @ Wo + bo  (f32 output)
// Grid (M/128, 1024/64), 256 threads. ctx is already f16 -> direct A frags.
// ---------------------------------------------------------------------------
__global__ __launch_bounds__(256) void out_gemm(
    const _Float16* __restrict__ Ctx, const float* __restrict__ Wo,
    const float* __restrict__ bo, float* __restrict__ out) {
  __shared__ _Float16 Bs[64 * LDSK];

  const int tid  = threadIdx.x;
  const int lane = tid & 31;
  const int w    = tid >> 5;
  const int lh   = lane & 15;
  const int hs   = lane >> 4;
  const int wm   = (w & 3) * 32;
  const int wn   = (w >> 2) * 32;
  const int m0   = blockIdx.x * 128;
  const int n0   = blockIdx.y * 64;

  v8f c[2][2] = {};

  for (int k0 = 0; k0 < DOUT; k0 += 32) {
#pragma unroll
    for (int p = 0; p < 2; ++p) {
      const int kk = (tid >> 4) + p * 16;
      const int nn = (tid & 15) * 4;
      const float4 v = *(const float4*)(Wo + (size_t)(k0 + kk) * DOUT + n0 + nn);
      Bs[(nn + 0) * LDSK + kk] = (_Float16)v.x;
      Bs[(nn + 1) * LDSK + kk] = (_Float16)v.y;
      Bs[(nn + 2) * LDSK + kk] = (_Float16)v.z;
      Bs[(nn + 3) * LDSK + kk] = (_Float16)v.w;
    }
    if (k0 + 32 < DOUT)
      __builtin_prefetch(Wo + (size_t)(k0 + 32 + (tid >> 4)) * DOUT + n0, 0, 0);
    __syncthreads();

    const _Float16* arow = Ctx + (size_t)(m0 + wm + lh) * DOUT + k0;
    const v16h a0 = frag_ld(arow, hs);
    const v16h a1 = frag_ld(arow + (size_t)16 * DOUT, hs);
    const v16h b0 = frag_ld(Bs + (wn + 0 + lh) * LDSK, hs);
    const v16h b1 = frag_ld(Bs + (wn + 16 + lh) * LDSK, hs);
    c[0][0] = wmma16(a0, b0, c[0][0]);
    c[0][1] = wmma16(a0, b1, c[0][1]);
    c[1][0] = wmma16(a1, b0, c[1][0]);
    c[1][1] = wmma16(a1, b1, c[1][1]);
    __syncthreads();
  }

#pragma unroll
  for (int i = 0; i < 2; ++i) {
#pragma unroll
    for (int j = 0; j < 2; ++j) {
#pragma unroll
      for (int r = 0; r < 8; ++r) {
        const int gi = m0 + wm + i * 16 + hs * 8 + r;
        const int gj = n0 + wn + j * 16 + lh;
        out[(size_t)gi * DOUT + gj] = c[i][j][r] + bo[gj];
      }
    }
  }
}

// ---------------------------------------------------------------------------
extern "C" void kernel_launch(void* const* d_in, const int* in_sizes, int n_in,
                              void* d_out, int out_size, void* d_ws,
                              size_t ws_size, hipStream_t stream) {
  (void)in_sizes; (void)n_in; (void)out_size; (void)ws_size;
  const float* x  = (const float*)d_in[0];
  const float* Wq = (const float*)d_in[1];
  const float* Wk = (const float*)d_in[2];
  const float* Wv = (const float*)d_in[3];
  const float* Wo = (const float*)d_in[4];
  const float* bo = (const float*)d_in[5];
  float* out = (float*)d_out;

  // f16 workspace: Q, K, Vt, ctx  (4 x 8 MB = 32 MB)
  const size_t QKV_ELEMS = (size_t)BDIM_B * NHEAD * SEQ * HDIM;  // 4,194,304
  _Float16* Qb  = (_Float16*)d_ws;
  _Float16* Kb  = Qb + QKV_ELEMS;
  _Float16* Vtb = Kb + QKV_ELEMS;
  _Float16* Ctx = Vtb + QKV_ELEMS;

  // 1) QKV projection: M=4096, N=3*1024, K=1024
  qkv_gemm<<<dim3((BDIM_B * SEQ) / 128, (3 * DOUT) / 64), 256, 0, stream>>>(
      x, Wq, Wk, Wv, Qb, Kb, Vtb);

  // 2) causal attention: B*H*QTILES waves / 8 waves per block
  attn_kernel<<<dim3((BDIM_B * NHEAD * QTILES) / 8), 256, 0, stream>>>(
      Qb, Kb, Vtb, Ctx);

  // 3) output projection: M=4096, N=1024, K=1024
  out_gemm<<<dim3((BDIM_B * SEQ) / 128, DOUT / 64), 256, 0, stream>>>(
      Ctx, Wo, bo, out);
}
